// ClassifierWAuto_69879117906562
// MI455X (gfx1250) — matrix operand order repaired and checked by reading, not compiled, not run on hardware
//
#include <hip/hip_runtime.h>

typedef __attribute__((ext_vector_type(16))) _Float16 v16h;
typedef __attribute__((ext_vector_type(8)))  float    v8f;

#define MSG_WAVES 4
#define MSG_BLOCKS_X 128   // grid-stride blocks per relation

// ---------------------------------------------------------------------------
// WMMA fragment layouts per CDNA5 ISA 7.12.2 (wave32, 16x16x32 f16)
// A 16x32 (MxK): lane l -> row M = l&15; K-half = l>>4.
//   VGPR v in 0..3 : K = 8*half + 2v + lo/hi
//   VGPR v in 4..7 : K = 16 + 8*half + 2(v-4) + lo/hi
// B 32x16 (KxN): lane l -> col N = l&15; VGPR v: same K mapping as A above
// C/D 16x16 f32: VGPR v: lanes0-15 M=v, lanes16-31 M=8+v; N = l&15
// ---------------------------------------------------------------------------
__device__ __forceinline__ v16h load_a_frag_f32(const float* __restrict__ Arow,
                                                int k0, int half, bool valid) {
  v16h a;
#pragma unroll
  for (int e = 0; e < 16; ++e) {
    int v = e >> 1, odd = e & 1;
    int k = (v < 4) ? (8 * half + 2 * v + odd)
                    : (16 + 8 * half + 2 * (v - 4) + odd);
    a[e] = valid ? (_Float16)Arow[k0 + k] : (_Float16)0.0f;
  }
  return a;
}

// ---------------------------------------------------------------------------
// Pack a [K,N] f32 weight matrix into fragment-major f16:
//   P[((kt*Ntiles + nt)*32 + lane)*16 + e] = W[kt*32 + kof(lane,e), nt*16 + (lane&15)]
// so each lane's v16h B-fragment is one contiguous, 32B-aligned chunk.
// ---------------------------------------------------------------------------
__global__ void pack_b_frag(const float* __restrict__ W, _Float16* __restrict__ P,
                            int K, int N) {
  long total = (long)K * N;
  long i = (long)blockIdx.x * blockDim.x + threadIdx.x;
  long stride = (long)gridDim.x * blockDim.x;
  int Ntiles = N >> 4;
  for (; i < total; i += stride) {
    int e = (int)(i & 15);
    int l = (int)((i >> 4) & 31);
    long f = i >> 9;
    int kt = (int)(f / Ntiles);
    int nt = (int)(f % Ntiles);
    int half = l >> 4, idx = l & 15;
    int v = e >> 1, odd = e & 1;
    int k = kt * 32 + ((v < 4) ? (8 * half + 2 * v + odd)
                               : (16 + 8 * half + 2 * (v - 4) + odd));
    int n = nt * 16 + idx;
    P[i] = (_Float16)W[(size_t)k * N + n];
  }
}

// ---------------------------------------------------------------------------
// GEMM: Cout[M,N] = act( A_f32[M,K] @ B_frag_f16[K,N] + bias + (Cadd?) )
// block = 256 threads = 8 waves, one 16-row strip; wave w -> 16-col tile.
// REQUIRES: N % 128 == 0, K % 32 == 0 (all call sites satisfy this).
// Wave 0 converts the shared A fragment and broadcasts it via LDS.
// ---------------------------------------------------------------------------
__global__ void wmma_gemm_bias_act(const float* __restrict__ A,
                                   const _Float16* __restrict__ Bp,
                                   const float* __restrict__ bias,
                                   const float* __restrict__ Cadd,
                                   float* __restrict__ Cout,
                                   int M, int N, int K, int relu) {
  __shared__ v16h aShr[32];
  int lane = threadIdx.x & 31;
  int wave = threadIdx.x >> 5;
  int half = lane >> 4, idx = lane & 15;
  int m0 = blockIdx.y * 16;
  int ntile = blockIdx.x * 8 + wave;      // 16-col tile index
  int n0 = ntile * 16;
  int Ntiles = N >> 4;
  int Ktiles = K >> 5;

  int arow_i = m0 + idx; if (arow_i >= M) arow_i = M - 1;   // tail clamp (read)
  const float* Arow = A + (size_t)arow_i * K;

  v8f c;
#pragma unroll
  for (int v = 0; v < 8; ++v) {
    int row = m0 + v + 8 * half; if (row >= M) row = M - 1;
    c[v] = Cadd ? Cadd[(size_t)row * N + (n0 + idx)] : 0.0f;
  }

  for (int kt = 0; kt < Ktiles; ++kt) {
    if (wave == 0)
      aShr[lane] = load_a_frag_f32(Arow, kt * 32, half, true);
    __syncthreads();
    v16h a = aShr[lane];
    const _Float16* bptr =
        Bp + (((size_t)kt * Ntiles + ntile) * 32 + lane) * 16;
    if (kt + 1 < Ktiles)
      __builtin_prefetch(Bp + (((size_t)(kt + 1) * Ntiles + ntile) * 32 + lane) * 16,
                         0, 1);
    v16h b = *(const v16h*)bptr;
    c = __builtin_amdgcn_wmma_f32_16x16x32_f16(false, a, false, b, (short)0, c,
                                               false, false);
    __syncthreads();
  }

#pragma unroll
  for (int v = 0; v < 8; ++v) {
    int row = m0 + v + 8 * half;
    if (row < M) {
      float val = c[v] + bias[n0 + idx];
      if (relu) val = fmaxf(val, 0.0f);
      Cout[(size_t)row * N + (n0 + idx)] = val;
    }
  }
}

// ---------------------------------------------------------------------------
// Edge bucketing by relation: histogram -> scan -> scatter
// ---------------------------------------------------------------------------
__global__ void zero_f32(float* p, long n) {
  long i = (long)blockIdx.x * blockDim.x + threadIdx.x;
  long stride = (long)gridDim.x * blockDim.x;
  for (; i < n; i += stride) p[i] = 0.0f;
}

__global__ void edge_hist(const int* __restrict__ rel, int* __restrict__ hist, int E) {
  int i = blockIdx.x * blockDim.x + threadIdx.x;
  if (i < E) atomicAdd(&hist[rel[i]], 1);
}

__global__ void edge_scan(const int* __restrict__ hist, int* __restrict__ offs,
                          int* __restrict__ curs, int R) {
  if (blockIdx.x == 0 && threadIdx.x == 0) {
    int acc = 0;
    for (int r = 0; r < R; ++r) { offs[r] = acc; curs[r] = acc; acc += hist[r]; }
    offs[R] = acc;
  }
}

__global__ void edge_scatter(const int* __restrict__ rel, int* __restrict__ curs,
                             int* __restrict__ eidx, int E) {
  int i = blockIdx.x * blockDim.x + threadIdx.x;
  if (i < E) {
    int pos = atomicAdd(&curs[rel[i]], 1);
    eidx[pos] = i;
  }
}

// ---------------------------------------------------------------------------
// Relational message kernel, grid = (MSG_BLOCKS_X, R), block = 128 = 4 waves.
// Packed W[r] (32 KB) is staged into LDS ONCE per block with
// GLOBAL_LOAD_ASYNC_TO_LDS_B128 (ASYNCcnt), then each wave grid-strides over
// groups of 16 same-relation edges: msg[16,128] = x[src,:] @ W[r] via WMMA,
// rows scatter-added into agg[dst,:] with global f32 atomics.
// ---------------------------------------------------------------------------
__global__ void msg_wmma_scatter(const float* __restrict__ x,
                                 const _Float16* __restrict__ Wp,
                                 const int* __restrict__ eidx,
                                 const int* __restrict__ offs,
                                 const int* __restrict__ src,
                                 const int* __restrict__ dst,
                                 float* __restrict__ agg, int H) {
  __shared__ __align__(32) _Float16 Wlds[128 * 128];   // 32 KB
  int r = blockIdx.y;
  int start = offs[r], end = offs[r + 1];
  // block-uniform: if even wave 0 of this block has no first group, all idle
  if (start + (int)blockIdx.x * (MSG_WAVES * 16) >= end) return;

  // ---- async-stage packed W[r] into LDS (16B chunks, all lanes) ----
  const _Float16* W = Wp + (size_t)r * H * H;
  {
    unsigned ldsbase = (unsigned)(size_t)(&Wlds[0]);   // low 32 bits = LDS offset
    int chunks = (H * H) / 8;                          // 16B per chunk
    for (int j = threadIdx.x; j < chunks; j += blockDim.x) {
      unsigned laddr = ldsbase + (unsigned)j * 16u;
      unsigned long long gaddr =
          (unsigned long long)(size_t)((const char*)W + (size_t)j * 16u);
      asm volatile("global_load_async_to_lds_b128 %0, %1, off"
                   :: "v"(laddr), "v"(gaddr) : "memory");
    }
    asm volatile("s_wait_asynccnt 0x0" ::: "memory");
  }
  __syncthreads();   // W[r] resident in LDS for the whole block

  int wave = threadIdx.x >> 5, lane = threadIdx.x & 31;
  int half = lane >> 4, idx = lane & 15;
  int gstride = (int)gridDim.x * MSG_WAVES * 16;

  for (int base = start + ((int)blockIdx.x * MSG_WAVES + wave) * 16;
       base < end; base += gstride) {
    int p = base + idx;
    bool valid = p < end;
    int e = valid ? eidx[p] : 0;
    int s = valid ? src[e] : 0;
    int d = valid ? dst[e] : -1;
    const float* Arow = x + (size_t)s * H;

    v16h a[4];
#pragma unroll
    for (int kk = 0; kk < 4; ++kk)
      a[kk] = load_a_frag_f32(Arow, kk * 32, half, valid);

    for (int nt = 0; nt < 8; ++nt) {
      v8f c = {};
#pragma unroll
      for (int kk = 0; kk < 4; ++kk) {
        const v16h b = *(const v16h*)(Wlds + (((kk * 8 + nt) * 32 + lane) * 16));
        c = __builtin_amdgcn_wmma_f32_16x16x32_f16(false, a[kk], false, b,
                                                   (short)0, c, false, false);
      }
      int col = nt * 16 + idx;
#pragma unroll
      for (int v = 0; v < 8; ++v) {
        int m = v + 8 * half;
        int dm = __shfl(d, m, 32);               // dst of output row m
        if (dm >= 0) atomicAdd(&agg[(size_t)dm * H + col], c[v]);
      }
    }
  }
}

// ---------------------------------------------------------------------------
// Per-graph mean pooling (graph_ids sorted) via atomics, then classifier.
// ---------------------------------------------------------------------------
__global__ void pool_sum(const float* __restrict__ x, const int* __restrict__ gid,
                         float* __restrict__ sums, int N, int H) {
  long i = (long)blockIdx.x * blockDim.x + threadIdx.x;
  long total = (long)N * H;
  if (i < total) {
    int node = (int)(i / H), col = (int)(i % H);
    atomicAdd(&sums[(size_t)gid[node] * H + col], x[i]);
  }
}

__global__ void pool_count(const int* __restrict__ gid, float* __restrict__ counts, int N) {
  int i = blockIdx.x * blockDim.x + threadIdx.x;
  if (i < N) atomicAdd(&counts[gid[i]], 1.0f);
}

__global__ void classify_softmax(const float* __restrict__ sums,
                                 const float* __restrict__ counts,
                                 const float* __restrict__ clsW,
                                 const float* __restrict__ clsb,
                                 float* __restrict__ out, int H, int C) {
  int g = blockIdx.x;
  __shared__ float logits[32];
  int t = threadIdx.x;
  float cnt = fmaxf(counts[g], 1.0f);
  if (t < C) {
    float acc = clsb[t];
    for (int k = 0; k < H; ++k)
      acc += (sums[(size_t)g * H + k] / cnt) * clsW[k * C + t];
    logits[t] = acc;
  }
  __syncthreads();
  if (t == 0) {
    float mx = -1e30f;
    for (int c = 0; c < C; ++c) mx = fmaxf(mx, logits[c]);
    float sum = 0.0f, ex[16];
    for (int c = 0; c < C; ++c) { ex[c] = __expf(logits[c] - mx); sum += ex[c]; }
    for (int c = 0; c < C; ++c) out[(size_t)g * C + c] = ex[c] / sum;
  }
}

// ---------------------------------------------------------------------------
// Host orchestration
// ---------------------------------------------------------------------------
static inline size_t alignUp(size_t x) { return (x + 255) & ~(size_t)255; }

extern "C" void kernel_launch(void* const* d_in, const int* in_sizes, int n_in,
                              void* d_out, int out_size, void* d_ws, size_t ws_size,
                              hipStream_t stream) {
  const float* h      = (const float*)d_in[0];
  const float* enc_W  = (const float*)d_in[1];
  const float* enc_b  = (const float*)d_in[2];
  const float* dec_W  = (const float*)d_in[3];
  const float* dec_b  = (const float*)d_in[4];
  const float* W1     = (const float*)d_in[5];
  const float* loop1  = (const float*)d_in[6];
  const float* b1     = (const float*)d_in[7];
  const float* W2     = (const float*)d_in[8];
  const float* loop2  = (const float*)d_in[9];
  const float* b2     = (const float*)d_in[10];
  const float* cls_W  = (const float*)d_in[11];
  const float* cls_b  = (const float*)d_in[12];
  const int*   src    = (const int*)d_in[13];
  const int*   dst    = (const int*)d_in[14];
  const int*   rel    = (const int*)d_in[15];
  const int*   gid    = (const int*)d_in[16];

  const int N  = in_sizes[16];
  const int E  = in_sizes[13];
  const int H  = in_sizes[2];
  const int IN = in_sizes[0] / N;
  const int R  = in_sizes[5] / (H * H);
  const int C  = in_sizes[12];
  const int G  = (out_size - N * IN) / C;

  // workspace carve-up
  char* ws = (char*)d_ws;
  size_t off = 0;
  auto take = [&](size_t bytes) -> void* {
    void* p = ws + off; off = alignUp(off + bytes); return p;
  };
  float*    xa     = (float*)take((size_t)N * H * 4);
  float*    xb     = (float*)take((size_t)N * H * 4);
  float*    agg    = (float*)take((size_t)N * H * 4);
  _Float16* p_enc  = (_Float16*)take((size_t)IN * H * 2);
  _Float16* p_dec  = (_Float16*)take((size_t)H * IN * 2);
  _Float16* p_w1   = (_Float16*)take((size_t)R * H * H * 2);
  _Float16* p_l1   = (_Float16*)take((size_t)H * H * 2);
  _Float16* p_w2   = (_Float16*)take((size_t)R * H * H * 2);
  _Float16* p_l2   = (_Float16*)take((size_t)H * H * 2);
  int*      hist   = (int*)take((size_t)R * 4);
  int*      offs   = (int*)take((size_t)(R + 1) * 4);
  int*      curs   = (int*)take((size_t)R * 4);
  int*      eidx   = (int*)take((size_t)E * 4);
  float*    sums   = (float*)take((size_t)G * H * 4);
  float*    counts = (float*)take((size_t)G * 4);
  (void)ws_size; (void)n_in;

  float* out_dec   = (float*)d_out;
  float* out_probs = (float*)d_out + (size_t)N * IN;

  // ---- pack weights into fragment-major f16 ----
  pack_b_frag<<<128, 256, 0, stream>>>(enc_W, p_enc, IN, H);
  pack_b_frag<<<128, 256, 0, stream>>>(dec_W, p_dec, H, IN);
  for (int r = 0; r < R; ++r) {
    pack_b_frag<<<64, 256, 0, stream>>>(W1 + (size_t)r * H * H,
                                        p_w1 + (size_t)r * H * H, H, H);
    pack_b_frag<<<64, 256, 0, stream>>>(W2 + (size_t)r * H * H,
                                        p_w2 + (size_t)r * H * H, H, H);
  }
  pack_b_frag<<<64, 256, 0, stream>>>(loop1, p_l1, H, H);
  pack_b_frag<<<64, 256, 0, stream>>>(loop2, p_l2, H, H);

  const int mrows = (N + 15) / 16;

  // ---- encoder: xa = relu(h @ enc_W + enc_b) ----
  wmma_gemm_bias_act<<<dim3(H / 128, mrows), 256, 0, stream>>>(
      h, p_enc, enc_b, nullptr, xa, N, H, IN, 1);

  // ---- decoder: d_out[0:N*IN] = xa @ dec_W + dec_b ----
  wmma_gemm_bias_act<<<dim3(IN / 128, mrows), 256, 0, stream>>>(
      xa, p_dec, dec_b, nullptr, out_dec, N, IN, H, 0);

  // ---- bucket edges by relation ----
  zero_f32<<<1, 64, 0, stream>>>((float*)hist, R);
  edge_hist<<<(E + 255) / 256, 256, 0, stream>>>(rel, hist, E);
  edge_scan<<<1, 1, 0, stream>>>(hist, offs, curs, R);
  edge_scatter<<<(E + 255) / 256, 256, 0, stream>>>(rel, curs, eidx, E);

  dim3 mgrid(MSG_BLOCKS_X, R);

  // ---- layer 1: xb = relu(agg + xa @ loop1 + b1) ----
  zero_f32<<<2048, 256, 0, stream>>>(agg, (long)N * H);
  msg_wmma_scatter<<<mgrid, MSG_WAVES * 32, 0, stream>>>(xa, p_w1, eidx, offs, src,
                                                         dst, agg, H);
  wmma_gemm_bias_act<<<dim3(1, mrows), 256, 0, stream>>>(
      xa, p_l1, b1, agg, xb, N, H, H, 1);

  // ---- layer 2: xa = relu(agg + xb @ loop2 + b2) ----
  zero_f32<<<2048, 256, 0, stream>>>(agg, (long)N * H);
  msg_wmma_scatter<<<mgrid, MSG_WAVES * 32, 0, stream>>>(xb, p_w2, eidx, offs, src,
                                                         dst, agg, H);
  wmma_gemm_bias_act<<<dim3(1, mrows), 256, 0, stream>>>(
      xb, p_l2, b2, agg, xa, N, H, H, 1);

  // ---- mean pooling + classifier ----
  zero_f32<<<256, 256, 0, stream>>>(sums, (long)G * H);
  zero_f32<<<8, 256, 0, stream>>>(counts, G);
  pool_sum<<<(int)(((long)N * H + 255) / 256), 256, 0, stream>>>(xa, gid, sums, N, H);
  pool_count<<<(N + 255) / 256, 256, 0, stream>>>(gid, counts, N);
  classify_softmax<<<G, 32, 0, stream>>>(sums, counts, cls_W, cls_b, out_probs, H, C);
}
